// MM_FiBiNET_71167608094992
// MI455X (gfx1250) — matrix operand order repaired and checked by reading, not compiled, not run on hardware
//
#include <hip/hip_runtime.h>
#include <hip/hip_bf16.h>

// ---------------------------------------------------------------------------
// Types / helpers
// ---------------------------------------------------------------------------
typedef __attribute__((ext_vector_type(16))) __bf16 v16bf;
typedef __attribute__((ext_vector_type(8)))  float  v8f;

union V16 {
    v16bf v;
    uint4 q[2];
};

// gfx1250 async global->LDS staging (guarded: falls back cleanly if the
// toolchain doesn't declare the builtins). Prototype (from hipcc diagnostic):
//   arg0: v4i (int __attribute__((vector_size(16)))) pointer, global AS
//   arg1: v4i pointer, LDS AS
//   arg2: int offset (imm), arg3: int cpol (imm)
#if __has_builtin(__builtin_amdgcn_global_load_async_to_lds_b128) && \
    __has_builtin(__builtin_amdgcn_s_wait_asynccnt)
#define ATHENA_ASYNC_LDS 1
typedef int v4i_t __attribute__((vector_size(16)));
typedef __attribute__((address_space(1))) v4i_t* glb_v4i_ptr;
typedef __attribute__((address_space(3))) v4i_t* lds_v4i_ptr;
#endif

__device__ __forceinline__ unsigned short f32_to_bf16(float f) {
    unsigned int u = __float_as_uint(f);
    unsigned int r = u + 0x7FFFu + ((u >> 16) & 1u);   // round-to-nearest-even
    return (unsigned short)(r >> 16);
}
__device__ __forceinline__ float bf16_to_f32(unsigned short h) {
    return __uint_as_float(((unsigned int)h) << 16);
}

// A-fragment (16x32 bf16): lane L holds row (L&15); kgrp=(L>>4) selects the
// K interleave half: V0-3 <- K=kgrp*8..+7, V4-7 <- K=16+kgrp*8..+7.
__device__ __forceinline__ v16bf ld_tile_a(const unsigned short* rowp, int k0, int kgrp) {
    V16 t;
    t.q[0] = *(const uint4*)(rowp + k0 + kgrp * 8);
    t.q[1] = *(const uint4*)(rowp + k0 + 16 + kgrp * 8);
    return t.v;
}
// Same fragment but sourced from an LDS-resident 16x32 tile (row stride 32).
__device__ __forceinline__ v16bf ld_tile_a_lds(const unsigned short* rowp, int kgrp) {
    V16 t;
    t.q[0] = *(const uint4*)(rowp + kgrp * 8);
    t.q[1] = *(const uint4*)(rowp + 16 + kgrp * 8);
    return t.v;
}
// B-fragment (32x16 bf16) from transposed weights Wt[n][k]: lane L holds
// column (L&15) = row of Wt; V0-3 <- K=kgrp*8..+7, V4-7 <- K=16+kgrp*8..+7.
__device__ __forceinline__ v16bf ld_tile_b(const unsigned short* rowp, int k0, int kgrp) {
    V16 t;
    t.q[0] = *(const uint4*)(rowp + k0 + kgrp * 8);
    t.q[1] = *(const uint4*)(rowp + k0 + 16 + kgrp * 8);
    return t.v;
}

// ---------------------------------------------------------------------------
// Weight convert / transpose to bf16:  dst[c][r] = src[r][c]
// ---------------------------------------------------------------------------
__global__ void transpose_cvt_bf16(const float* __restrict__ src,
                                   unsigned short* __restrict__ dst,
                                   int rows, int cols) {
    int idx = blockIdx.x * blockDim.x + threadIdx.x;
    if (idx >= rows * cols) return;
    int r = idx / cols, c = idx % cols;
    dst[(long)c * rows + r] = f32_to_bf16(src[idx]);
}

__global__ void convert_bf16_kernel(const float* __restrict__ src,
                                    unsigned short* __restrict__ dst, long n) {
    long idx = (long)blockIdx.x * blockDim.x + threadIdx.x;
    if (idx < n) dst[idx] = f32_to_bf16(src[idx]);
}

// ---------------------------------------------------------------------------
// Generic bf16 WMMA GEMM:  out[M,N] = A[M,K] * Wt[N,K]^T  (+ epilogue)
// Block: 256 threads = 8 waves; tile = 16 rows x 128 cols (16 per wave).
// A row r lives at A + (r/rowDiv)*sOuter + (r%rowDiv)*sInner.
// mode 0: store bf16 raw; mode 1: +bias, store f32;
// mode 2: +bias, BN(running stats)+ReLU, store bf16.
// STAGE_A=true: the 16x32 A-chunk is DMA'd into a double-buffered LDS tile
// with GLOBAL_LOAD_ASYNC_TO_LDS_B128 (waves 0-1 issue; ASYNCcnt-tracked),
// removing the 8x redundant per-wave A loads. Falls back to direct loads if
// the async builtins are unavailable.
// ---------------------------------------------------------------------------
template <bool STAGE_A>
__global__ void gemm_bf16_wmma(const unsigned short* __restrict__ A,
                               int rowDiv, int sOuter, int sInner,
                               const unsigned short* __restrict__ Wt,
                               int K, int M,
                               float* __restrict__ outF,
                               unsigned short* __restrict__ outH,
                               int ldo, int mode,
                               const float* __restrict__ bias,
                               const float* __restrict__ g,
                               const float* __restrict__ beta,
                               const float* __restrict__ rm,
                               const float* __restrict__ rv) {
    const int tid  = threadIdx.x;
    const int wave = tid >> 5;
    const int lane = tid & 31;
    const int kgrp = lane >> 4;
    const int l16  = lane & 15;

    const int m0 = blockIdx.x * 16;
    const int n0 = blockIdx.y * 128 + wave * 16;

    int arow = m0 + l16;
    if (arow >= M) arow = M - 1;
    const unsigned short* ap = A + (long)(arow / rowDiv) * sOuter
                                 + (long)(arow % rowDiv) * sInner;
    const unsigned short* bp = Wt + (long)(n0 + l16) * K;

    v8f acc = {0.f, 0.f, 0.f, 0.f, 0.f, 0.f, 0.f, 0.f};

#ifdef ATHENA_ASYNC_LDS
    if constexpr (STAGE_A) {
        // double-buffered 16x32 bf16 A tile: 2 x 1 KB in LDS
        __shared__ unsigned short tileA[2][16 * 32];
        const int lrow = tid >> 2;            // loader row (tid < 64)
        const int lseg = tid & 3;             // 4 x 16B segments per row
        int grow = m0 + lrow;
        if (grow >= M) grow = M - 1;
        const unsigned short* gap = A + (long)(grow / rowDiv) * sOuter
                                      + (long)(grow % rowDiv) * sInner
                                      + lseg * 8;

        // prologue: fill buffer 0
        if (tid < 64) {
            __builtin_amdgcn_global_load_async_to_lds_b128(
                (glb_v4i_ptr)gap,
                (lds_v4i_ptr)&tileA[0][lrow * 32 + lseg * 8], 0, 0);
        }
        __builtin_amdgcn_s_wait_asynccnt(0);
        __syncthreads();

        int buf = 0;
        for (int k0 = 0; k0 < K; k0 += 32) {
            // overlap: DMA next chunk into the other buffer while computing
            if (tid < 64 && (k0 + 32) < K) {
                __builtin_amdgcn_global_load_async_to_lds_b128(
                    (glb_v4i_ptr)(gap + k0 + 32),
                    (lds_v4i_ptr)&tileA[buf ^ 1][lrow * 32 + lseg * 8], 0, 0);
            }
            v16bf af = ld_tile_a_lds(&tileA[buf][l16 * 32], kgrp);
            v16bf bf = ld_tile_b(bp, k0, kgrp);
            acc = __builtin_amdgcn_wmma_f32_16x16x32_bf16(
                false, af, false, bf, (short)0, acc, false, false);
            __builtin_amdgcn_s_wait_asynccnt(0);
            __syncthreads();
            buf ^= 1;
        }
    } else
#endif
    {
        for (int k0 = 0; k0 < K; k0 += 32) {
            v16bf af = ld_tile_a(ap, k0, kgrp);
            v16bf bf = ld_tile_b(bp, k0, kgrp);
            acc = __builtin_amdgcn_wmma_f32_16x16x32_bf16(
                false, af, false, bf, (short)0, acc, false, false);
        }
    }

    // C layout: VGPR r -> (m = m0 + r + 8*kgrp, n = n0 + l16)
    const int n = n0 + l16;
    float bi = 0.f, sc = 0.f, sh = 0.f;
    if (mode >= 1) bi = bias[n];
    if (mode == 2) {
        float inv = rsqrtf(rv[n] + 1e-5f);
        sc = g[n] * inv;
        sh = beta[n] - rm[n] * sc;
    }
#pragma unroll
    for (int r = 0; r < 8; ++r) {
        int m = m0 + r + kgrp * 8;
        float v = acc[r] + bi;
        if (mode == 2) v = fmaxf(v * sc + sh, 0.f);
        long o = (long)m * ldo + n;
        if (mode == 1) outF[o] = v;
        else           outH[o] = f32_to_bf16(v);
    }
}

// ---------------------------------------------------------------------------
// Per-sample feature kernel: LN+ReLU on mm, gathers, history pooling, SENet.
// Writes xs (bf16) into c[:, 0:768]. 1 block / sample, 128 threads.
// ---------------------------------------------------------------------------
__global__ void features_kernel(const int* __restrict__ item_id,
                                const int* __restrict__ likes,
                                const int* __restrict__ views,
                                const int* __restrict__ item_seq,
                                const float* __restrict__ item_emb,
                                const float* __restrict__ cate_emb,
                                const float* __restrict__ mm_raw,
                                const float* __restrict__ ln_g,
                                const float* __restrict__ ln_b,
                                const float* __restrict__ se_W1,
                                const float* __restrict__ se_b1,
                                const float* __restrict__ se_W2,
                                const float* __restrict__ se_b2,
                                unsigned short* __restrict__ c_bf) {
    const int b = blockIdx.x;
    const int e = threadIdx.x;           // 0..127
    __shared__ float red[128];
    __shared__ float zsh[6];
    __shared__ float wsh[6];

    // ---- LayerNorm + ReLU over mm_raw[b, :] ----
    float mmv = mm_raw[(long)b * 128 + e];
    red[e] = mmv; __syncthreads();
    for (int s = 64; s > 0; s >>= 1) { if (e < s) red[e] += red[e + s]; __syncthreads(); }
    float mu = red[0] * (1.f / 128.f); __syncthreads();
    float d = mmv - mu;
    red[e] = d * d; __syncthreads();
    for (int s = 64; s > 0; s >>= 1) { if (e < s) red[e] += red[e + s]; __syncthreads(); }
    float var = red[0] * (1.f / 128.f); __syncthreads();
    float mm_act = fmaxf(d * rsqrtf(var + 1e-5f) * ln_g[e] + ln_b[e], 0.f);

    // ---- gathers ----
    float like = cate_emb[(long)likes[b] * 128 + e];
    float view = cate_emb[(long)views[b] * 128 + e];
    int iid = item_id[b];
    float item = (iid == 0) ? 0.f : item_emb[(long)iid * 128 + e];

    // ---- masked-mean history pooling ----
    float hs = 0.f; int cnt = 0;
    for (int t = 0; t < 50; ++t) {
        int s = item_seq[(long)b * 50 + t];
        if (s != 0) { hs += item_emb[(long)s * 128 + e]; ++cnt; }
    }
    float hist = hs / (float)(cnt > 0 ? cnt : 1);

    float x[6] = {0.f, like, view, item, mm_act, hist};

    // ---- SENet: z = mean_e(x), w = sigmoid(relu(z@W1+b1)@W2+b2) ----
    for (int f = 0; f < 6; ++f) {
        red[e] = x[f]; __syncthreads();
        for (int s = 64; s > 0; s >>= 1) { if (e < s) red[e] += red[e + s]; __syncthreads(); }
        if (e == 0) zsh[f] = red[0] * (1.f / 128.f);
        __syncthreads();
    }
    if (e == 0) {
        float h1[3];
        for (int j = 0; j < 3; ++j) {
            float s = se_b1[j];
            for (int f = 0; f < 6; ++f) s += zsh[f] * se_W1[f * 3 + j];
            h1[j] = fmaxf(s, 0.f);
        }
        for (int f = 0; f < 6; ++f) {
            float s = se_b2[f];
            for (int j = 0; j < 3; ++j) s += h1[j] * se_W2[j * 6 + f];
            wsh[f] = 1.f / (1.f + __expf(-s));
        }
    }
    __syncthreads();

    const long base = (long)b * 2688;
    for (int f = 0; f < 6; ++f)
        c_bf[base + f * 128 + e] = f32_to_bf16(x[f] * wsh[f]);
}

// ---------------------------------------------------------------------------
// Pair products: c[:, 768 + p*128 + e] = xs[:, I[p], e] * xw[:, J[p], e]
// ---------------------------------------------------------------------------
__global__ void pairs_kernel(const unsigned short* __restrict__ xw_bf,
                             unsigned short* __restrict__ c_bf) {
    const int IP[15] = {0,0,0,0,0,1,1,1,1,2,2,2,3,3,4};
    const int JP[15] = {1,2,3,4,5,2,3,4,5,3,4,5,4,5,5};
    int b = blockIdx.x / 15;
    int p = blockIdx.x % 15;
    int e = threadIdx.x;
    long cb = (long)b * 2688;
    float xi = bf16_to_f32(c_bf[cb + IP[p] * 128 + e]);
    float xj = bf16_to_f32(xw_bf[((long)b * 6 + JP[p]) * 128 + e]);
    c_bf[cb + 768 + p * 128 + e] = f32_to_bf16(xi * xj);
}

// ---------------------------------------------------------------------------
// Final: dot(h2[b,:], W3) + b3 -> sigmoid. 1 block / sample, 256 threads.
// ---------------------------------------------------------------------------
__global__ void final_kernel(const unsigned short* __restrict__ h2,
                             const float* __restrict__ W3,
                             const float* __restrict__ b3,
                             float* __restrict__ out) {
    int b = blockIdx.x;
    int t = threadIdx.x;
    __shared__ float red[256];
    red[t] = bf16_to_f32(h2[(long)b * 256 + t]) * W3[t];
    __syncthreads();
    for (int s = 128; s > 0; s >>= 1) { if (t < s) red[t] += red[t + s]; __syncthreads(); }
    if (t == 0) out[b] = 1.f / (1.f + __expf(-(red[0] + b3[0])));
}

// ---------------------------------------------------------------------------
// Host launcher
// ---------------------------------------------------------------------------
extern "C" void kernel_launch(void* const* d_in, const int* in_sizes, int n_in,
                              void* d_out, int out_size, void* d_ws, size_t ws_size,
                              hipStream_t stream) {
    constexpr int B = 16384;

    const int*   item_id  = (const int*)  d_in[0];
    const float* emb128   = (const float*)d_in[1];
    const int*   likes    = (const int*)  d_in[2];
    const int*   views    = (const int*)  d_in[3];
    const int*   item_seq = (const int*)  d_in[4];
    const float* item_emb = (const float*)d_in[5];
    const float* cate_emb = (const float*)d_in[6];
    const float* mm_W     = (const float*)d_in[7];
    const float* mm_b     = (const float*)d_in[8];
    const float* ln_g     = (const float*)d_in[9];
    const float* ln_b     = (const float*)d_in[10];
    const float* se_W1    = (const float*)d_in[11];
    const float* se_b1    = (const float*)d_in[12];
    const float* se_W2    = (const float*)d_in[13];
    const float* se_b2    = (const float*)d_in[14];
    const float* bi_W     = (const float*)d_in[15];
    const float* mlp_W1   = (const float*)d_in[16];
    const float* mlp_b1   = (const float*)d_in[17];
    const float* bn1_g    = (const float*)d_in[18];
    const float* bn1_b    = (const float*)d_in[19];
    const float* bn1_rm   = (const float*)d_in[20];
    const float* bn1_rv   = (const float*)d_in[21];
    const float* mlp_W2   = (const float*)d_in[22];
    const float* mlp_b2   = (const float*)d_in[23];
    const float* bn2_g    = (const float*)d_in[24];
    const float* bn2_b    = (const float*)d_in[25];
    const float* bn2_rm   = (const float*)d_in[26];
    const float* bn2_rv   = (const float*)d_in[27];
    const float* mlp_W3   = (const float*)d_in[28];
    const float* mlp_b3   = (const float*)d_in[29];
    float* out = (float*)d_out;

    // -------- workspace layout (all offsets 256B aligned) --------
    char* ws = (char*)d_ws;
    unsigned short* mm_Wt  = (unsigned short*)(ws + 0);          //  32 KB
    unsigned short* bi_Wt  = (unsigned short*)(ws + 32768);      //  32 KB
    unsigned short* W1t    = (unsigned short*)(ws + 65536);      // 2.75 MB (512 x 2688)
    unsigned short* W2t    = (unsigned short*)(ws + 2818048);    // 256 KB  (256 x 512)
    unsigned short* a0_bf  = (unsigned short*)(ws + 3080192);    // B*128 bf16
    float*          mm_raw = (float*)         (ws + 7274496);    // B*128 f32
    unsigned short* c_bf   = (unsigned short*)(ws + 15663104);   // B*2688 bf16
    unsigned short* xw_bf  = (unsigned short*)(ws + 103743488);  // B*6*128 bf16
    unsigned short* h1_bf  = (unsigned short*)(ws + 128909312);  // B*512 bf16
    unsigned short* h2_bf  = (unsigned short*)(ws + 145686528);  // B*256 bf16

    // 1) weights -> bf16, transposed (Wt[n][k])
    transpose_cvt_bf16<<<(128 * 128 + 255) / 256, 256, 0, stream>>>(mm_W, mm_Wt, 128, 128);
    transpose_cvt_bf16<<<(128 * 128 + 255) / 256, 256, 0, stream>>>(bi_W, bi_Wt, 128, 128);
    transpose_cvt_bf16<<<(2688 * 512 + 255) / 256, 256, 0, stream>>>(mlp_W1, W1t, 2688, 512);
    transpose_cvt_bf16<<<(512 * 256 + 255) / 256, 256, 0, stream>>>(mlp_W2, W2t, 512, 256);
    convert_bf16_kernel<<<(B * 128 + 255) / 256, 256, 0, stream>>>(emb128, a0_bf, (long)B * 128);

    // 2) mm projection: (B x 128) @ (128 x 128) + bias -> f32
    gemm_bf16_wmma<false><<<dim3(B / 16, 1), 256, 0, stream>>>(
        a0_bf, 1, 128, 0, mm_Wt, 128, B,
        mm_raw, nullptr, 128, 1, mm_b, nullptr, nullptr, nullptr, nullptr);

    // 3) features: LN+ReLU, gathers, history mean, SENet -> xs into c[:, :768]
    features_kernel<<<B, 128, 0, stream>>>(
        item_id, likes, views, item_seq, item_emb, cate_emb, mm_raw,
        ln_g, ln_b, se_W1, se_b1, se_W2, se_b2, c_bf);

    // 4) bilinear: (B*6 x 128) @ (128 x 128) -> xw (bf16)
    gemm_bf16_wmma<false><<<dim3((B * 6) / 16, 1), 256, 0, stream>>>(
        c_bf, 6, 2688, 128, bi_Wt, 128, B * 6,
        nullptr, xw_bf, 128, 0, nullptr, nullptr, nullptr, nullptr, nullptr);

    // 5) pair products -> c[:, 768:]
    pairs_kernel<<<B * 15, 128, 0, stream>>>(xw_bf, c_bf);

    // 6) MLP1: (B x 2688) @ (2688 x 512) + bias + BN + ReLU -> h1 (bf16)
    //    (async LDS-staged A tile: 8x less redundant A traffic)
    gemm_bf16_wmma<true><<<dim3(B / 16, 4), 256, 0, stream>>>(
        c_bf, 1, 2688, 0, W1t, 2688, B,
        nullptr, h1_bf, 512, 2, mlp_b1, bn1_g, bn1_b, bn1_rm, bn1_rv);

    // 7) MLP2: (B x 512) @ (512 x 256) + bias + BN + ReLU -> h2 (bf16)
    gemm_bf16_wmma<true><<<dim3(B / 16, 2), 256, 0, stream>>>(
        h1_bf, 1, 512, 0, W2t, 512, B,
        nullptr, h2_bf, 256, 2, mlp_b2, bn2_g, bn2_b, bn2_rm, bn2_rv);

    // 8) final dot + sigmoid
    final_kernel<<<B, 256, 0, stream>>>(h2_bf, mlp_W3, mlp_b3, out);

    (void)in_sizes; (void)n_in; (void)out_size; (void)ws_size;
}
